// CCModel_71545565217305
// MI455X (gfx1250) — matrix operand order repaired and checked by reading, not compiled, not run on hardware
//
#include <hip/hip_runtime.h>

typedef __attribute__((ext_vector_type(16))) _Float16 v16h;
typedef __attribute__((ext_vector_type(2)))  _Float16 v2h;
typedef __attribute__((ext_vector_type(8)))  float    v8f;
typedef unsigned int u32x4 __attribute__((ext_vector_type(4)));
typedef int          i32x8 __attribute__((ext_vector_type(8)));
typedef int          i32x4 __attribute__((ext_vector_type(4)));

#define NBATCH 32
#define NCH    128
#define NFREQ  2049
#define NROW   (NBATCH * NCH)      // 4096 rows
#define KDIM   4128                // 2*2049 = 4098 padded to multiple of 32
#define KH     (KDIM / 2)          // 2064 half2 per A row
#define NLAGS  599                 // 2*NLAG-1
#define NPAD   640                 // padded to multiple of 64 (BN)
#define NLAG   300
#define NMA    20
#define NK     (KDIM / 32)         // 129 K-steps

// workspace layout (bytes)
#define A_OFF 0
#define B_OFF ((size_t)NROW * KDIM * 2)                 // 33,816,576
#define X_OFF (B_OFF + (size_t)NPAD * KDIM * 2)         // + 5,283,840

// ---------------------------------------------------------------------------
// TDM: issue a 2D tensor_load_to_lds. Tile is tile_dim1 rows x 32 halves,
// row stride KDIM halves in global; LDS gets +16B pad after each 64B row
// (pad_interval=3 -> 16 DWORDs, pad_amount=3 -> 4 DWORDs) => LDS row = 40
// halves, matching the fragment loader's LDT.
// (6-arg builtin form: this toolchain is clang-23 / therock-10 HIP headers.)
// ---------------------------------------------------------------------------
__device__ __forceinline__ void tdm_load_2d(unsigned lds_off, const void* gptr,
                                            int tensor_dim1, int tile_dim1) {
  unsigned long long ga = (unsigned long long)(size_t)gptr;
  u32x4 g0;
  g0[0] = 1u;                                              // count=1
  g0[1] = lds_off;                                         // lds_addr (bytes)
  g0[2] = (unsigned)(ga & 0xffffffffu);                    // global_addr lo
  g0[3] = (unsigned)((ga >> 32) & 0x01ffffffu) | (2u << 30); // addr hi | type=2
  i32x8 g1;
  g1[0] = (1 << 16) | (1 << 20) | (3 << 22) | (3 << 25);   // 2B, pad 64B->+16B
  g1[1] = (KDIM & 0xffff) << 16;                           // tensor_dim0 lo16
  g1[2] = (KDIM >> 16) | ((tensor_dim1 & 0xffff) << 16);   // dim0 hi | dim1 lo
  g1[3] = (tensor_dim1 >> 16) | (32 << 16);                // dim1 hi | tile_dim0
  g1[4] = tile_dim1;                                       // tile_dim1, dim2=0
  g1[5] = KDIM;                                            // dim0 stride lo
  g1[6] = 0;
  g1[7] = 0;
  i32x4 z4 = {};
  i32x8 z8 = {};
  __builtin_amdgcn_tensor_load_to_lds(g0, g1, z4, z4, z8, 0);
}

// ---------------------------------------------------------------------------
// Phase 1: cross-spectrum -> f16 A[4096][4128], scaled 1/64, zero K tail.
// ---------------------------------------------------------------------------
__global__ void build_a(const float2* __restrict__ d1,
                        const float2* __restrict__ d2,
                        _Float16* __restrict__ A) {
  int gid = blockIdx.x * blockDim.x + threadIdx.x;
  if (gid >= NROW * KH) return;
  int row = gid / KH;
  int j   = gid - row * KH;
  v2h out = {};
  if (j < NFREQ) {
    float2 x1 = d1[(size_t)row * NFREQ + j];
    float2 x2 = d2[(size_t)row * NFREQ + j];
    float re = x1.x * x2.x + x1.y * x2.y;   // Re(conj(c1)*c2)
    float im = x1.x * x2.y - x1.y * x2.x;   // Im(conj(c1)*c2)
    out[0] = (_Float16)(re * (1.0f / 64.0f));
    out[1] = (_Float16)(im * (1.0f / 64.0f));
  }
  *(v2h*)(A + (size_t)row * KDIM + 2 * j) = out;
}

// ---------------------------------------------------------------------------
// Phase 2: TRANSPOSED trig table BtT[NPAD][KDIM] (f16).
// BtT[t][2k]   =  (a_k/64) * cos(2*pi*k*n/4096)   n = t - 299
// BtT[t][2k+1] = -(a_k/64) * sin(2*pi*k*n/4096)   a_k = 1 at k=0,2048 else 2
// ---------------------------------------------------------------------------
__global__ void build_b(_Float16* __restrict__ BtT) {
  int gid = blockIdx.x * blockDim.x + threadIdx.x;
  if (gid >= NPAD * KDIM) return;
  int t  = gid / KDIM;
  int k2 = gid - t * KDIM;
  int k  = k2 >> 1;
  float val = 0.0f;
  if (k < NFREQ && t < NLAGS) {
    int n = t - (NLAG - 1);                 // -299 .. +299
    int p = (k * n) % 4096;
    if (p < 0) p += 4096;
    float th = (float)p * 1.5339807878856412e-3f;   // 2*pi/4096
    float s, c;
    __sincosf(th, &s, &c);
    float a = ((k == 0) || (k == NFREQ - 1)) ? (1.0f / 64.0f) : (2.0f / 64.0f);
    val = (k2 & 1) ? (-a * s) : (a * c);
  }
  BtT[gid] = (_Float16)val;
}

// ---------------------------------------------------------------------------
// Phase 3: WMMA GEMM  X[4096][640] = A[4096][4128] * BtT^T
// 256 threads = 8 waves; BM=128, BN=64; tiles double-buffered in LDS and
// delivered by the Tensor Data Mover (wave 0 issues, TENSORcnt sync).
// One barrier per K-step; 4 x v_wmma_f32_16x16x32_f16 per wave per step.
// ---------------------------------------------------------------------------
__global__ __launch_bounds__(256)
void xcorr_gemm(const _Float16* __restrict__ A,
                const _Float16* __restrict__ BtT,
                float* __restrict__ X) {
  constexpr int LDT = 40;                // halves per LDS row (64B + 16B pad)
  constexpr int ASZ = 128 * LDT;         // A tile halves
  constexpr int BSZ = 64 * LDT;          // B tile halves
  __shared__ _Float16 lds[2 * (ASZ + BSZ)];

  const int tid  = threadIdx.x;
  const int lane = tid & 31;
  const int w    = tid >> 5;
  const int bm0  = blockIdx.x * 128;
  const int bn0  = blockIdx.y * 64;

  const unsigned lds_base = (unsigned)(size_t)(void*)&lds[0];
  const _Float16* agp = A   + (size_t)bm0 * KDIM;   // A tile col 0 of this row strip
  const _Float16* bgp = BtT + (size_t)bn0 * KDIM;   // B tile (n-major) col 0

  v8f acc[4];
#pragma unroll
  for (int j = 0; j < 4; ++j) acc[j] = (v8f){};

  if (w == 0) {
    tdm_load_2d(lds_base,            agp, NROW, 128);   // A tile, buf 0
    tdm_load_2d(lds_base + ASZ * 2,  bgp, NPAD, 64);    // B tile, buf 0
  }

  for (int kt = 0; kt < NK; ++kt) {
    const int buf = kt & 1;
    if (w == 0) __builtin_amdgcn_s_wait_tensorcnt(0);
    __syncthreads();        // tiles(kt) visible; everyone done with other buf
    if (w == 0 && kt + 1 < NK) {
      const unsigned nb = lds_base + (unsigned)((buf ^ 1) * (ASZ + BSZ)) * 2;
      tdm_load_2d(nb,           agp + (kt + 1) * 32, NROW, 128);
      tdm_load_2d(nb + ASZ * 2, bgp + (kt + 1) * 32, NPAD, 64);
    }

    const _Float16* As = &lds[buf * (ASZ + BSZ)];
    const _Float16* Bs = As + ASZ;

    // A fragment: lanes 0-15 M=0..15 K{0..7,16..23}; lanes 16-31 K{8..15,24..31}
    v16h af;
    {
      const int m  = lane & 15;
      const int kb = (lane >> 4) * 8;
      const _Float16* ap = &As[(16 * w + m) * LDT];
#pragma unroll
      for (int v = 0; v < 8; ++v) {
        int k = (v < 4) ? (kb + 2 * v) : (16 + kb + 2 * (v - 4));
        v2h p = *(const v2h*)(ap + k);
        af[2 * v] = p[0]; af[2 * v + 1] = p[1];
      }
    }
    // B fragments (LDS holds B transposed [n][k]): lanes 0-15 K{0..7,16..23},
    // lanes 16-31 K{8..15,24..31}, 2 K per VGPR.
    const int nn  = lane & 15;
    const int kb2 = (lane >> 4) * 8;
#pragma unroll
    for (int j = 0; j < 4; ++j) {
      v16h bf;
      const _Float16* bp = &Bs[(nn + 16 * j) * LDT + kb2];
#pragma unroll
      for (int v = 0; v < 8; ++v) {
        v2h p = *(const v2h*)(bp + ((v < 4) ? (2 * v) : (16 + 2 * (v - 4))));
        bf[2 * v] = p[0]; bf[2 * v + 1] = p[1];
      }
      acc[j] = __builtin_amdgcn_wmma_f32_16x16x32_f16(false, af, false, bf,
                                                      (short)0, acc[j],
                                                      false, false);
    }
  }

  // D layout: vgpr r, lane L -> n = L&15, m = r + 8*(L>>4)
  const int n   = lane & 15;
  const int mhi = (lane >> 4) * 8;
#pragma unroll
  for (int j = 0; j < 4; ++j)
#pragma unroll
    for (int r = 0; r < 8; ++r) {
      int row = bm0 + 16 * w + r + mhi;
      X[(size_t)row * NPAD + bn0 + 16 * j + n] = acc[j][r];
    }
}

// ---------------------------------------------------------------------------
// Phase 4: channel moving-average (window 20, clip == zero-pad) + peak pick.
// ---------------------------------------------------------------------------
__global__ __launch_bounds__(128)
void smooth_reduce(const float* __restrict__ X, float* __restrict__ out) {
  __shared__ float sx[128], sn[128];
  __shared__ int   ix[128], in_[128];
  const int bc  = blockIdx.x;
  const int b   = bc >> 7;
  const int c   = bc & 127;
  const int tid = threadIdx.x;
  const int clo = (c - NMA / 2 < 0) ? 0 : (c - NMA / 2);
  const int chi = (c + NMA / 2 - 1 > NCH - 1) ? (NCH - 1) : (c + NMA / 2 - 1);

  float mv = -1e30f, nv = 1e30f;
  int   mi = 0,      ni = 0;
  for (int l = tid; l < NLAGS; l += 128) {
    float s = 0.0f;
    for (int cw = clo; cw <= chi; ++cw)
      s += X[(size_t)(b * NCH + cw) * NPAD + l];
    s *= (1.0f / NMA);
    if (s > mv) { mv = s; mi = l; }
    if (s < nv) { nv = s; ni = l; }
  }
  sx[tid] = mv; ix[tid] = mi; sn[tid] = nv; in_[tid] = ni;
  __syncthreads();
  for (int off = 64; off > 0; off >>= 1) {
    if (tid < off) {
      float ov = sx[tid + off]; int oi = ix[tid + off];
      if (ov > sx[tid] || (ov == sx[tid] && oi < ix[tid])) { sx[tid] = ov; ix[tid] = oi; }
      float uv = sn[tid + off]; int ui = in_[tid + off];
      if (uv < sn[tid] || (uv == sn[tid] && ui < in_[tid])) { sn[tid] = uv; in_[tid] = ui; }
    }
    __syncthreads();
  }
  if (tid == 0) {
    float vmax = sx[0], vmin = sn[0];
    bool ineg = fabsf(vmin) > vmax;
    out[bc]        = ineg ? vmin : vmax;
    int id         = ineg ? in_[0] : ix[0];
    out[NROW + bc] = (float)(id - (NLAG - 1)) * 0.01f;   // tshift
  }
}

// ---------------------------------------------------------------------------
// Phase 5: events pass-through (value-cast int32 -> float32 slots).
// ---------------------------------------------------------------------------
__global__ void copy_events(const int* __restrict__ e1,
                            const int* __restrict__ e2,
                            float* __restrict__ out) {
  int t = threadIdx.x;
  if (t < NBATCH)          out[2 * NROW + t]               = (float)e1[t];
  else if (t < 2 * NBATCH) out[2 * NROW + NBATCH + t - 32] = (float)e2[t - 32];
}

extern "C" void kernel_launch(void* const* d_in, const int* in_sizes, int n_in,
                              void* d_out, int out_size, void* d_ws, size_t ws_size,
                              hipStream_t stream) {
  const float2* d1 = (const float2*)d_in[0];
  const float2* d2 = (const float2*)d_in[1];
  const int*    e1 = (const int*)d_in[2];
  const int*    e2 = (const int*)d_in[3];
  float* out = (float*)d_out;
  char*  ws  = (char*)d_ws;

  _Float16* A   = (_Float16*)(ws + A_OFF);
  _Float16* BtT = (_Float16*)(ws + B_OFF);
  float*    X   = (float*)(ws + X_OFF);

  {
    int total = NROW * KH;
    build_a<<<(total + 255) / 256, 256, 0, stream>>>(d1, d2, A);
  }
  {
    int total = NPAD * KDIM;
    build_b<<<(total + 255) / 256, 256, 0, stream>>>(BtT);
  }
  {
    dim3 grid(NROW / 128, NPAD / 64);
    xcorr_gemm<<<grid, dim3(256), 0, stream>>>(A, BtT, X);
  }
  smooth_reduce<<<NROW, 128, 0, stream>>>(X, out);
  copy_events<<<1, 64, 0, stream>>>(e1, e2, out);
}
